// Decoder_45999099740779
// MI455X (gfx1250) — matrix operand order repaired and checked by reading, not compiled, not run on hardware
//
#include <hip/hip_runtime.h>

// ---------------- types ----------------
typedef __bf16 bf16_t;
typedef bf16_t v16bf __attribute__((ext_vector_type(16)));
typedef float  v8f   __attribute__((ext_vector_type(8)));

union FragAB {
    uint4 u[2];   // 32 bytes
    v16bf v;      // 16 x bf16
};

// ---------------- helpers ----------------
__device__ __forceinline__ unsigned short f2bf(float f) {
    unsigned u = __builtin_bit_cast(unsigned, f);
    unsigned r = u + 0x7FFFu + ((u >> 16) & 1u);   // round-to-nearest-even
    return (unsigned short)(r >> 16);
}
__device__ __forceinline__ float bf2f(unsigned short h) {
    unsigned u = ((unsigned)h) << 16;
    return __builtin_bit_cast(float, u);
}
__device__ __forceinline__ v8f splat8(float x) {
    v8f r;
#pragma unroll
    for (int i = 0; i < 8; ++i) r[i] = x;
    return r;
}
__device__ __forceinline__ float sig_f(float x) {
    return __builtin_amdgcn_rcpf(1.0f + __expf(-x));
}
__device__ __forceinline__ float tanh_f(float x) {
    float ax = __builtin_fabsf(x);
    float e  = __expf(-2.0f * ax);
    float t  = (1.0f - e) * __builtin_amdgcn_rcpf(1.0f + e);
    return __builtin_copysignf(t, x);
}

// ---------------- problem constants ----------------
#define BATCH   128
#define DDIM    512
#define HDIM    1024
#define TSTEPS  256
#define NGATES  4096          // 4*H
#define KGATES  1536          // D + H
#define KT_G    48            // KGATES/32
#define KT_FC   32            // HDIM/32
#define WAVES   16
#define MTILE   16            // batch rows per workgroup

// LDS layout (bytes). 16B pad per row spreads DS banks for b128 frag loads.
#define XT_STRIDE  (DDIM*2 + 16)      // 1040
#define H_STRIDE   (HDIM*2 + 16)      // 2064
#define OFF_XT     0
#define OFF_H0     (OFF_XT + MTILE*XT_STRIDE)            // 16640
#define OFF_H1     (OFF_H0 + MTILE*H_STRIDE)             // 49664
#define OFF_LNH    (OFF_H1 + MTILE*H_STRIDE)             // 82688
#define OFF_BIAS   (OFF_LNH + MTILE*H_STRIDE)            // 115712
#define OFF_GAM    (OFF_BIAS + NGATES*4)                 // 132096
#define OFF_BET    (OFF_GAM + HDIM*4)                    // 136192
#define OFF_BFC    (OFF_BET + HDIM*4)                    // 140288
#define LDS_BYTES  (OFF_BFC + DDIM*4)                    // 142336

// workspace layout (bytes)
#define WS_PW    0
#define WS_PWFC  (NGATES*KGATES*2)                       // 12,582,912
#define WS_BIAS  (WS_PWFC + DDIM*HDIM*2)                 // 13,631,488

// ---------------- prep kernels: pack weights into WMMA B-fragment order ----
// B frag layout per ISA 7.12.2 (16-bit B, 32x16): lane n(0..15): N=n, K=0..7,16..23
// packed 2/VGPR; lane 16+n: N=n, K=8..15,24..31.  We store each 16x32 tile as
// [lane][16 bf16] contiguous -> two global_load_b128 per lane at consume time.
__global__ void pack_gates_kernel(const float* __restrict__ Wih,
                                  const float* __restrict__ Whh,
                                  unsigned short* __restrict__ dst) {
    unsigned p    = blockIdx.x * 256u + threadIdx.x;     // exactly 4096*1536
    unsigned tile = p >> 9, r = p & 511u;
    unsigned lane = r >> 4, j = r & 15u;
    unsigned nt = tile / KT_G, kt = tile % KT_G;
    unsigned n = nt * 16u + (lane & 15u);
    unsigned k = kt * 32u + ((lane >> 4) << 3) + (j & 7u) + ((j >> 3) << 4);
    float v = (k < DDIM) ? Wih[(size_t)n * DDIM + k]
                         : Whh[(size_t)n * HDIM + (k - DDIM)];
    dst[p] = f2bf(v);
}

__global__ void pack_fc_kernel(const float* __restrict__ Wfc,
                               unsigned short* __restrict__ dst) {
    unsigned p    = blockIdx.x * 256u + threadIdx.x;     // exactly 512*1024
    unsigned tile = p >> 9, r = p & 511u;
    unsigned lane = r >> 4, j = r & 15u;
    unsigned nt = tile / KT_FC, kt = tile % KT_FC;
    unsigned n = nt * 16u + (lane & 15u);
    unsigned k = kt * 32u + ((lane >> 4) << 3) + (j & 7u) + ((j >> 3) << 4);
    dst[p] = f2bf(Wfc[(size_t)n * HDIM + k]);
}

__global__ void bias_kernel(const float* __restrict__ bih,
                            const float* __restrict__ bhh,
                            float* __restrict__ dst) {
    unsigned i = blockIdx.x * 256u + threadIdx.x;        // 4096
    dst[i] = bih[i] + bhh[i];
}

// ---------------- persistent LSTM decoder ----------------
// grid = 8 WGs (16 batch rows each), block = 512 threads = 16 wave32.
// Each wave owns a 64-col slice of each gate (4 n-tiles of 16), keeps that
// slice's cell state c in VGPRs across all 256 steps.
__global__ void __launch_bounds__(512, 1)
lstm_decoder_kernel(const float* __restrict__ x,
                    const unsigned short* __restrict__ pw,
                    const unsigned short* __restrict__ pwfc,
                    const float* __restrict__ biasw,
                    const float* __restrict__ gamma,
                    const float* __restrict__ beta,
                    const float* __restrict__ bfc,
                    float* __restrict__ out) {
    extern __shared__ char smem[];
    char*  XT   = smem + OFF_XT;
    char*  HB0  = smem + OFF_H0;
    char*  HB1  = smem + OFF_H1;
    char*  LNH  = smem + OFF_LNH;
    float* BIAS = (float*)(smem + OFF_BIAS);
    float* GAM  = (float*)(smem + OFF_GAM);
    float* BET  = (float*)(smem + OFF_BET);
    float* BFC  = (float*)(smem + OFF_BFC);

    const int tid    = threadIdx.x;
    const int wave   = tid >> 5;
    const int lane   = tid & 31;
    const int laneM  = lane & 15;     // row (A) or column (B/C) within tile
    const int laneHi = lane >> 4;     // K-half / M-half selector
    const int wgBase = blockIdx.x * MTILE;

    const uint4* pw4   = (const uint4*)pw;    // 64 uint4 per 16x32 tile
    const uint4* pwfc4 = (const uint4*)pwfc;

    // ---- stage constants + initial state into LDS ----
    for (int i = tid; i < NGATES; i += 512) BIAS[i] = biasw[i];
    for (int i = tid; i < HDIM;   i += 512) { GAM[i] = gamma[i]; BET[i] = beta[i]; }
    if (tid < DDIM) BFC[tid] = bfc[tid];
    for (int i = tid; i < MTILE * DDIM; i += 512) {
        int r = i >> 9, cix = i & (DDIM - 1);
        *(unsigned short*)(XT + r * XT_STRIDE + cix * 2) =
            f2bf(x[(size_t)(wgBase + r) * DDIM + cix]);
    }
    for (int i = tid; i < MTILE * HDIM; i += 512) {
        int r = i >> 10, cix = i & (HDIM - 1);
        *(unsigned short*)(HB0 + r * H_STRIDE + cix * 2) = 0;
    }
    __syncthreads();

    v8f cst[4];
#pragma unroll
    for (int tg = 0; tg < 4; ++tg) cst[tg] = splat8(0.0f);

    char* Hcur = HB0;
    char* Hnxt = HB1;

    for (int t = 0; t < TSTEPS; ++t) {
        // ================= Phase A/B: gates GEMM + cell update ============
#pragma unroll
        for (int tg = 0; tg < 4; ++tg) {
            const int cb = wave * 64 + tg * 16;                  // col in gate
            v8f ai = splat8(BIAS[          cb + laneM]);
            v8f af = splat8(BIAS[  HDIM  + cb + laneM]);
            v8f ag = splat8(BIAS[2*HDIM  + cb + laneM]);
            v8f ao = splat8(BIAS[3*HDIM  + cb + laneM]);
            const int ntb = wave * 4 + tg;                       // 0..63

            // K = 0..511 : A from xt buffer
#pragma unroll 2
            for (int kt = 0; kt < 16; ++kt) {
                FragAB a;
                const char* ab = XT + laneM * XT_STRIDE + kt * 64 + laneHi * 16;
                a.u[0] = *(const uint4*)ab;
                a.u[1] = *(const uint4*)(ab + 32);
                FragAB b0, b1, b2, b3;
                const uint4* q;
                q = pw4 + (((0 * 64 + ntb) * KT_G + kt) << 5) + (lane << 1);
                b0.u[0] = q[0]; b0.u[1] = q[1];
                q = pw4 + (((1 * 64 + ntb) * KT_G + kt) << 5) + (lane << 1);
                b1.u[0] = q[0]; b1.u[1] = q[1];
                q = pw4 + (((2 * 64 + ntb) * KT_G + kt) << 5) + (lane << 1);
                b2.u[0] = q[0]; b2.u[1] = q[1];
                q = pw4 + (((3 * 64 + ntb) * KT_G + kt) << 5) + (lane << 1);
                b3.u[0] = q[0]; b3.u[1] = q[1];
                ai = __builtin_amdgcn_wmma_f32_16x16x32_bf16(false, a.v, false, b0.v, (short)0, ai, false, false);
                af = __builtin_amdgcn_wmma_f32_16x16x32_bf16(false, a.v, false, b1.v, (short)0, af, false, false);
                ag = __builtin_amdgcn_wmma_f32_16x16x32_bf16(false, a.v, false, b2.v, (short)0, ag, false, false);
                ao = __builtin_amdgcn_wmma_f32_16x16x32_bf16(false, a.v, false, b3.v, (short)0, ao, false, false);
            }
            // K = 512..1535 : A from current h buffer
#pragma unroll 2
            for (int kt = 16; kt < KT_G; ++kt) {
                FragAB a;
                const char* ab = Hcur + laneM * H_STRIDE + (kt - 16) * 64 + laneHi * 16;
                a.u[0] = *(const uint4*)ab;
                a.u[1] = *(const uint4*)(ab + 32);
                FragAB b0, b1, b2, b3;
                const uint4* q;
                q = pw4 + (((0 * 64 + ntb) * KT_G + kt) << 5) + (lane << 1);
                b0.u[0] = q[0]; b0.u[1] = q[1];
                q = pw4 + (((1 * 64 + ntb) * KT_G + kt) << 5) + (lane << 1);
                b1.u[0] = q[0]; b1.u[1] = q[1];
                q = pw4 + (((2 * 64 + ntb) * KT_G + kt) << 5) + (lane << 1);
                b2.u[0] = q[0]; b2.u[1] = q[1];
                q = pw4 + (((3 * 64 + ntb) * KT_G + kt) << 5) + (lane << 1);
                b3.u[0] = q[0]; b3.u[1] = q[1];
                ai = __builtin_amdgcn_wmma_f32_16x16x32_bf16(false, a.v, false, b0.v, (short)0, ai, false, false);
                af = __builtin_amdgcn_wmma_f32_16x16x32_bf16(false, a.v, false, b1.v, (short)0, af, false, false);
                ag = __builtin_amdgcn_wmma_f32_16x16x32_bf16(false, a.v, false, b2.v, (short)0, ag, false, false);
                ao = __builtin_amdgcn_wmma_f32_16x16x32_bf16(false, a.v, false, b3.v, (short)0, ao, false, false);
            }
            // cell update entirely in registers; write new h (bf16) to Hnxt
#pragma unroll
            for (int e = 0; e < 8; ++e) {
                float iv = sig_f(ai[e]);
                float fv = sig_f(af[e]);
                float gv = tanh_f(ag[e]);
                float ov = sig_f(ao[e]);
                float cn = fv * cst[tg][e] + iv * gv;
                cst[tg][e] = cn;
                float hv = ov * tanh_f(cn);
                int row = e + laneHi * 8;                    // C-layout M
                *(unsigned short*)(Hnxt + row * H_STRIDE + (cb + laneM) * 2) = f2bf(hv);
            }
        }
        __syncthreads();

        // ================= Phase C: LayerNorm (wave w -> batch row w) =====
        {
            const char* rb = Hnxt + wave * H_STRIDE + lane * 64;  // 32 cols/lane
            float s = 0.f, ss = 0.f;
#pragma unroll
            for (int j = 0; j < 32; ++j) {
                float v = bf2f(*(const unsigned short*)(rb + j * 2));
                s += v; ss += v * v;
            }
            for (int off = 16; off > 0; off >>= 1) {
                s  += __shfl_xor(s,  off, 32);
                ss += __shfl_xor(ss, off, 32);
            }
            float mu  = s  * (1.0f / HDIM);
            float var = ss * (1.0f / HDIM) - mu * mu;
            float rs  = __builtin_amdgcn_rsqf(var + 1e-5f);
            int colb = lane * 32;
            char* ob = LNH + wave * H_STRIDE + colb * 2;
#pragma unroll
            for (int j = 0; j < 32; j += 2) {
                float v0 = bf2f(*(const unsigned short*)(rb + j * 2));
                float v1 = bf2f(*(const unsigned short*)(rb + j * 2 + 2));
                float o0 = (v0 - mu) * rs * GAM[colb + j]     + BET[colb + j];
                float o1 = (v1 - mu) * rs * GAM[colb + j + 1] + BET[colb + j + 1];
                *(unsigned int*)(ob + j * 2) =
                    (unsigned)f2bf(o0) | ((unsigned)f2bf(o1) << 16);
            }
        }
        __syncthreads();

        // ================= Phase D: FC GEMM, write y -> out and next xt ===
#pragma unroll
        for (int tt = 0; tt < 2; ++tt) {
            const int nfc  = wave * 2 + tt;      // 0..31
            const int colb = nfc * 16;
            v8f acc = splat8(BFC[colb + laneM]);
#pragma unroll 2
            for (int kt = 0; kt < KT_FC; ++kt) {
                FragAB a;
                const char* ab = LNH + laneM * H_STRIDE + kt * 64 + laneHi * 16;
                a.u[0] = *(const uint4*)ab;
                a.u[1] = *(const uint4*)(ab + 32);
                FragAB b;
                const uint4* q = pwfc4 + ((nfc * KT_FC + kt) << 5) + (lane << 1);
                b.u[0] = q[0]; b.u[1] = q[1];
                acc = __builtin_amdgcn_wmma_f32_16x16x32_bf16(false, a.v, false, b.v, (short)0, acc, false, false);
            }
#pragma unroll
            for (int e = 0; e < 8; ++e) {
                int row = e + laneHi * 8;
                int col = colb + laneM;
                out[((size_t)(wgBase + row) * TSTEPS + t) * DDIM + col] = acc[e];
                *(unsigned short*)(XT + row * XT_STRIDE + col * 2) = f2bf(acc[e]);
            }
        }
        __syncthreads();

        // swap h double-buffers
        char* tmp = Hcur; Hcur = Hnxt; Hnxt = tmp;
    }
}

// ---------------- launch ----------------
extern "C" void kernel_launch(void* const* d_in, const int* in_sizes, int n_in,
                              void* d_out, int out_size, void* d_ws, size_t ws_size,
                              hipStream_t stream) {
    (void)in_sizes; (void)n_in; (void)out_size; (void)ws_size;
    const float* x     = (const float*)d_in[0];
    // d_in[1] = target_input (unused by reference)
    const float* Wih   = (const float*)d_in[2];
    const float* Whh   = (const float*)d_in[3];
    const float* bih   = (const float*)d_in[4];
    const float* bhh   = (const float*)d_in[5];
    const float* gamma = (const float*)d_in[6];
    const float* beta  = (const float*)d_in[7];
    const float* Wfc   = (const float*)d_in[8];
    const float* bfc   = (const float*)d_in[9];
    float* out = (float*)d_out;

    char* ws = (char*)d_ws;
    unsigned short* pw    = (unsigned short*)(ws + WS_PW);
    unsigned short* pwfc  = (unsigned short*)(ws + WS_PWFC);
    float*          biasw = (float*)(ws + WS_BIAS);

    // prep: pack weights to bf16 WMMA-fragment tiles (L2-resident afterwards)
    pack_gates_kernel<<<(NGATES * KGATES) / 256, 256, 0, stream>>>(Wih, Whh, pw);
    pack_fc_kernel<<<(DDIM * HDIM) / 256, 256, 0, stream>>>(Wfc, pwfc);
    bias_kernel<<<NGATES / 256, 256, 0, stream>>>(bih, bhh, biasw);

    // persistent decoder: 8 WGs x 16 batch rows, full T-loop on-chip
    lstm_decoder_kernel<<<dim3(BATCH / MTILE), dim3(512), LDS_BYTES, stream>>>(
        x, pw, pwfc, biasw, gamma, beta, bfc, out);
}
